// DilatedCausalSelfAttention_83519934038696
// MI455X (gfx1250) — matrix operand order repaired and checked by reading, not compiled, not run on hardware
//
#include <hip/hip_runtime.h>

// ---------------------------------------------------------------------------
// Types for CDNA5 WMMA (wave32): V_WMMA_F32_16X16X32_BF16
// ---------------------------------------------------------------------------
typedef __bf16 bf16;
typedef __attribute__((ext_vector_type(16))) __bf16 v16bf;
typedef __attribute__((ext_vector_type(8)))  float  v8f;

static __device__ __forceinline__ v8f wmma_bf16(v16bf a, v16bf b, v8f c) {
  // (neg_a, A, neg_b, B, c_mod, C, reuse_a, reuse_b)
  return __builtin_amdgcn_wmma_f32_16x16x32_bf16(false, a, false, b,
                                                 (short)0, c, false, false);
}

static __device__ __forceinline__ bf16 f2bf(float f) {
  // round-to-nearest-even fp32 -> bf16
  unsigned u = __float_as_uint(f);
  unsigned r = (u + 0x7FFFu + ((u >> 16) & 1u)) >> 16;
  unsigned short s = (unsigned short)r;
  return *reinterpret_cast<bf16*>(&s);
}

union FragU { v16bf v; uint4 q[2]; };
union V8BF  { uint4 q; bf16 e[8]; };

// A-fragment (16x32 bf16, row M = lane%16):
//   lanes 0-15  : elems 0..7 -> K=0..7,   elems 8..15 -> K=16..23
//   lanes 16-31 : elems 0..7 -> K=8..15,  elems 8..15 -> K=24..31
static __device__ __forceinline__ v16bf load_frag_a(const bf16* base, int ld,
                                                    int row, int c0, int half) {
  FragU u;
  const bf16* pr = base + (size_t)row * ld + c0 + 8 * half;
  u.q[0] = *(const uint4*)(pr);
  u.q[1] = *(const uint4*)(pr + 16);
  return u.v;
}

// B-fragment (32x16 bf16, col N = lane%16):
//   lanes 0-15  : elems 0..7 -> K=0..7,   elems 8..15 -> K=16..23
//   lanes 16-31 : elems 0..7 -> K=8..15,  elems 8..15 -> K=24..31
static __device__ __forceinline__ v16bf load_frag_b(const bf16* base, int ld,
                                                    int row, int c0, int half) {
  FragU u;
  const bf16* pr = base + (size_t)row * ld + c0 + 8 * half;
  u.q[0] = *(const uint4*)(pr);
  u.q[1] = *(const uint4*)(pr + 16);
  return u.v;
}

// ---------------------------------------------------------------------------
// Problem constants
// ---------------------------------------------------------------------------
#define NP   16        // batch(4) * segments(4) independent problems
#define MSEQ 512       // dilated rows per problem
#define CDIM 1024
#define QKVN 3072
#define NHEAD 16
#define DH   64
#define VT_LD 520      // padded LDS pitch (bank-conflict-free transpose)

// ---------------------------------------------------------------------------
// Kernel 1: gather dilated rows (0::4 within each 2048 segment) + f32->bf16
// ---------------------------------------------------------------------------
__global__ void k_gather_x(const float* __restrict__ x, bf16* __restrict__ xi) {
  const int total = NP * MSEQ * CDIM;
  for (int idx = blockIdx.x * blockDim.x + threadIdx.x; idx < total;
       idx += gridDim.x * blockDim.x) {
    int m = idx >> 10, c = idx & 1023;
    int p = m >> 9, i = m & 511;
    size_t src = ((size_t)(p >> 2) * 8192 + (size_t)(p & 3) * 2048 +
                  (size_t)(i << 2)) * CDIM + c;
    xi[idx] = f2bf(x[src]);
  }
}

__global__ void k_cvt_bf16(const float* __restrict__ src, bf16* __restrict__ dst,
                           int n) {
  for (int i = blockIdx.x * blockDim.x + threadIdx.x; i < n;
       i += gridDim.x * blockDim.x)
    dst[i] = f2bf(src[i]);
}

__global__ void k_zero_f32(float* __restrict__ dst, int n) {
  float4 z = {0.f, 0.f, 0.f, 0.f};
  int n4 = n >> 2;
  for (int i = blockIdx.x * blockDim.x + threadIdx.x; i < n4;
       i += gridDim.x * blockDim.x)
    ((float4*)dst)[i] = z;
}

// ---------------------------------------------------------------------------
// Kernel 2/5: WMMA GEMM  Y[M,N] = X[M,K] * W[N,K]^T + bias[N]
//   block = 256 threads = 8 waves (4Mx2N), wave tile 32x64, block tile 128x128
//   SCATTER=false: write bf16, pitch ldy (QKV).
//   SCATTER=true : write f32 into dilated rows of (4,8192,1024) output.
// ---------------------------------------------------------------------------
template <bool SCATTER>
__global__ void __launch_bounds__(256)
k_wmma_gemm(const bf16* __restrict__ X, const bf16* __restrict__ W,
            const float* __restrict__ bias,
            bf16* __restrict__ Ybf, float* __restrict__ Yf,
            int Kdim, int ldy) {
  const int tid = threadIdx.x;
  const int w = tid >> 5, lane = tid & 31, r = lane & 15, half = lane >> 4;
  const int m0 = blockIdx.y * 128 + (w >> 1) * 32;
  const int n0 = blockIdx.x * 128 + (w & 1) * 64;

  v8f acc[2][4] = {};
  for (int c0 = 0; c0 < Kdim; c0 += 32) {
    v16bf a0 = load_frag_a(X, Kdim, m0 + r,      c0, half);
    v16bf a1 = load_frag_a(X, Kdim, m0 + 16 + r, c0, half);
#pragma unroll
    for (int Nf = 0; Nf < 4; ++Nf) {
      v16bf b = load_frag_b(W, Kdim, n0 + Nf * 16 + r, c0, half);
      acc[0][Nf] = wmma_bf16(a0, b, acc[0][Nf]);
      acc[1][Nf] = wmma_bf16(a1, b, acc[1][Nf]);
    }
  }

#pragma unroll
  for (int Nf = 0; Nf < 4; ++Nf) {
    int n = n0 + Nf * 16 + r;
    float bv = bias[n];
#pragma unroll
    for (int Mf = 0; Mf < 2; ++Mf) {
#pragma unroll
      for (int i = 0; i < 8; ++i) {
        int m = m0 + Mf * 16 + i + 8 * half;   // C/D layout: M = i + 8*half
        float val = acc[Mf][Nf][i] + bv;
        if (SCATTER) {
          int p = m >> 9, ii = m & 511;
          size_t orow = (size_t)(p >> 2) * 8192 + (size_t)(p & 3) * 2048 +
                        (size_t)(ii << 2);
          Yf[orow * CDIM + n] = val;
        } else {
          Ybf[(size_t)m * ldy + n] = f2bf(val);
        }
      }
    }
  }
}

// ---------------------------------------------------------------------------
// Kernel 3: causal flash attention per (problem p, head h)
//   block = 256 threads = 8 waves; 32 chunks of 16 query rows, wave w takes
//   chunks {2w, 2w+1, 30-2w, 31-2w} so every wave sees ~constant causal work.
//   S = Q K^T via WMMA, online softmax (max via shfl; SUM via ones-column
//   WMMA accumulator -> no shuffle for denominators), O += P V via WMMA.
//   LDS: Vt[80][VT_LD] bf16 (rows 64..79 == 1.0) + P[8][16][32] bf16
// ---------------------------------------------------------------------------
__global__ void __launch_bounds__(256)
k_attn(const bf16* __restrict__ qkv, bf16* __restrict__ obuf) {
  const int h = blockIdx.x, p = blockIdx.y;
  extern __shared__ char smem[];
  bf16* Vt = (bf16*)smem;                          // 80 x VT_LD
  bf16* Pl = (bf16*)(smem + 80 * VT_LD * 2);       // 8 x 16 x 32

  const int tid = threadIdx.x;
  const int w = tid >> 5, lane = tid & 31, r = lane & 15, half = lane >> 4;

  const size_t rowbase = (size_t)p * MSEQ * QKVN;
  const bf16* Q  = qkv + rowbase + h * DH;
  const bf16* Km = qkv + rowbase + CDIM + h * DH;
  const bf16* Vm = qkv + rowbase + 2 * CDIM + h * DH;

  // stage V transposed: Vt[d][k] = V[k][d], vectorized 8-wide global reads
  for (int idx = tid; idx < (MSEQ * DH) / 8; idx += 256) {
    int k  = idx >> 3;                 // V row
    int dc = (idx & 7) * 8;            // V col chunk of 8
    V8BF v;
    v.q = *(const uint4*)(Vm + (size_t)k * QKVN + dc);
#pragma unroll
    for (int j = 0; j < 8; ++j)
      Vt[(dc + j) * VT_LD + k] = v.e[j];
  }
  {
    unsigned short ob = 0x3F80;                    // bf16 1.0
    bf16 one = *(bf16*)&ob;
    for (int idx = tid; idx < 16 * MSEQ; idx += 256)
      Vt[(DH + (idx >> 9)) * VT_LD + (idx & 511)] = one;
  }
  __syncthreads();

  bf16* Pw = Pl + w * 16 * 32;
  const int chunkid[4] = {2 * w, 2 * w + 1, 30 - 2 * w, 31 - 2 * w};

  for (int cc = 0; cc < 4; ++cc) {
    const int qb = chunkid[cc] * 16;               // query-row base, 16 rows

    v16bf qa[2];
#pragma unroll
    for (int kk = 0; kk < 2; ++kk)
      qa[kk] = load_frag_a(Q, QKVN, qb + r, kk * 32, half);

    v8f o[4] = {};
    v8f osum = {};                                 // softmax denominators
    float mrun[8];
#pragma unroll
    for (int i = 0; i < 8; ++i) mrun[i] = -1e30f;

    for (int kt = 0; kt < qb + 16; kt += 32) {     // causal: keys <= qb+15
      v8f s[2] = {};
#pragma unroll
      for (int kk = 0; kk < 2; ++kk) {
        v16bf b0 = load_frag_b(Km, QKVN, kt + r,      kk * 32, half);
        v16bf b1 = load_frag_b(Km, QKVN, kt + 16 + r, kk * 32, half);
        s[0] = wmma_bf16(qa[kk], b0, s[0]);
        s[1] = wmma_bf16(qa[kk], b1, s[1]);
      }

      // scale 1/sqrt(64), causal mask, online max + rescale
#pragma unroll
      for (int i = 0; i < 8; ++i) {
        int row = qb + i + 8 * half;
        float v0 = s[0][i] * 0.125f;
        float v1 = s[1][i] * 0.125f;
        if (kt + r      > row) v0 = -1e30f;
        if (kt + 16 + r > row) v1 = -1e30f;
        float tm = fmaxf(v0, v1);
#pragma unroll
        for (int off = 1; off < 16; off <<= 1)
          tm = fmaxf(tm, __shfl_xor(tm, off, 32));
        float mnew = fmaxf(mrun[i], tm);
        float alpha = __expf(mrun[i] - mnew);
        mrun[i] = mnew;
        float p0 = __expf(v0 - mnew);
        float p1 = __expf(v1 - mnew);
#pragma unroll
        for (int Nf = 0; Nf < 4; ++Nf)
          o[Nf][i] = o[Nf][i] * alpha;
        osum[i] = osum[i] * alpha;
        // stage P (C-layout) into LDS for A-layout reload
        Pw[(i + 8 * half) * 32 + r]      = f2bf(p0);
        Pw[(i + 8 * half) * 32 + 16 + r] = f2bf(p1);
      }

      // reload P as one 16x32 A-fragment (same wave; compiler inserts waits)
      v16bf pa = load_frag_a(Pw, 32, r, 0, half);
#pragma unroll
      for (int Nf = 0; Nf < 4; ++Nf) {
        v16bf vb = load_frag_b(Vt, VT_LD, Nf * 16 + r, kt, half);
        o[Nf] = wmma_bf16(pa, vb, o[Nf]);
      }
      // denominator: l += P @ ones  (Vt rows 64..79 are 1.0)
      v16bf vones = load_frag_b(Vt, VT_LD, DH + r, kt, half);
      osum = wmma_bf16(pa, vones, osum);
    }

    // epilogue: normalize and store bf16 into Obuf[8192,1024]
#pragma unroll
    for (int i = 0; i < 8; ++i) {
      float inv = 1.0f / osum[i];
      size_t grow = (size_t)p * MSEQ + qb + i + 8 * half;
#pragma unroll
      for (int Nf = 0; Nf < 4; ++Nf)
        obuf[grow * CDIM + h * DH + Nf * 16 + r] = f2bf(o[Nf][i] * inv);
    }
  }
}

// ---------------------------------------------------------------------------
// Host launcher
// ---------------------------------------------------------------------------
extern "C" void kernel_launch(void* const* d_in, const int* in_sizes, int n_in,
                              void* d_out, int out_size, void* d_ws,
                              size_t ws_size, hipStream_t stream) {
  (void)in_sizes; (void)n_in;
  const float* x     = (const float*)d_in[0];
  const float* w_in  = (const float*)d_in[1];
  const float* b_in  = (const float*)d_in[2];
  const float* w_out = (const float*)d_in[3];
  const float* b_out = (const float*)d_in[4];
  float* out = (float*)d_out;

  char* ws = (char*)d_ws;
  size_t off = 0;
  bf16* xi_bf   = (bf16*)(ws + off); off += (size_t)NP * MSEQ * CDIM * 2;
  bf16* w_in_bf = (bf16*)(ws + off); off += (size_t)QKVN * CDIM * 2;
  bf16* w_out_bf= (bf16*)(ws + off); off += (size_t)CDIM * CDIM * 2;
  bf16* qkv     = (bf16*)(ws + off); off += (size_t)NP * MSEQ * QKVN * 2;
  bf16* obuf    = (bf16*)(ws + off); off += (size_t)NP * MSEQ * CDIM * 2;
  if (off > ws_size) return;  // workspace too small; nothing safe to do

  // 1) gather dilated rows + convert operands to bf16
  k_gather_x<<<4096, 256, 0, stream>>>(x, xi_bf);
  k_cvt_bf16<<<3072, 256, 0, stream>>>(w_in, w_in_bf, QKVN * CDIM);
  k_cvt_bf16<<<1024, 256, 0, stream>>>(w_out, w_out_bf, CDIM * CDIM);

  // 2) QKV projection: [8192,1024] x [3072,1024]^T + b_in -> bf16 [8192,3072]
  k_wmma_gemm<false><<<dim3(QKVN / 128, (NP * MSEQ) / 128), 256, 0, stream>>>(
      xi_bf, w_in_bf, b_in, qkv, nullptr, CDIM, QKVN);

  // 3) causal flash attention, one block per (head, problem)
  k_attn<<<dim3(NHEAD, NP), 256, 80 * VT_LD * 2 + 8 * 16 * 32 * 2, stream>>>(
      qkv, obuf);

  // 4) zero full output (non-dilated rows stay zero)
  k_zero_f32<<<4096, 256, 0, stream>>>(out, out_size);

  // 5) output projection + bias, scattered into rows 0::4 of each segment
  k_wmma_gemm<true><<<dim3(CDIM / 128, (NP * MSEQ) / 128), 256, 0, stream>>>(
      obuf, w_out_bf, b_out, nullptr, out, CDIM, 0);
}
